// PermutohedralLayer_77446850281692
// MI455X (gfx1250) — matrix operand order repaired and checked by reading, not compiled, not run on hardware
//
#include <hip/hip_runtime.h>

// ---------------- problem constants (fixed by reference) ----------------
#define DD    4
#define NVX   5            // DD+1 simplex vertices
#define NB    4            // batches
#define NPT   1024         // points per batch (H*W = 32*32)
#define CIN   64
#define COUT  64
#define M0    (NPT*NVX)    // 5120 lattice entries per batch
#define FNB   31           // neighbor taps (2^(D+1)-1, includes self)
#define HS    16384        // hash slots per batch (power of 2, ~31% load)
#define HMASK (HS-1)
#define KTOT  (FNB*CIN)    // 1984
#define KT    (KTOT/32)    // 62 wmma K-steps
#define MTILES (M0/16)     // 320 M-tiles per batch

typedef __attribute__((ext_vector_type(16))) __bf16 v16bf;
typedef __attribute__((ext_vector_type(8)))  float  v8f;
typedef __attribute__((ext_vector_type(2)))  int    v2i;
typedef __attribute__((address_space(1))) v2i* gv2i_p;   // global b64 chunk
typedef __attribute__((address_space(3))) v2i* lv2i_p;   // LDS b64 chunk

__device__ __forceinline__ unsigned int hashc(int code) {
    return ((unsigned int)code * 2654435761u) & HMASK;
}
__device__ __forceinline__ unsigned short f2bf(float f) {
    unsigned int u = __float_as_uint(f);
    u = (u + 0x7FFFu + ((u >> 16) & 1u)) >> 16;   // round-to-nearest-even
    return (unsigned short)u;
}

// ---------------- init: clear hash/table state every call ----------------
__global__ void k_init(int* hashKey, int* hashDense, int* nUniq, float* table,
                       unsigned short* tableBF) {
    int t = blockIdx.x * blockDim.x + threadIdx.x;
    if (t < NB * M0 * CIN) table[t] = 0.0f;
    if (t < NB * HS) { hashKey[t] = -1; hashDense[t] = -1; }
    if (t < NB) nUniq[t] = 0;
    // zero pad row used by async A-tile staging for invalid neighbors
    if (t < CIN) tableBF[(size_t)NB * M0 * CIN + t] = 0;
}

// ---------------- per-point: elevate, rank, barycentric, vertex codes ----
__global__ void k_points(const float* __restrict__ features,
                         float* __restrict__ bary, int* __restrict__ codes) {
    int t = blockIdx.x * blockDim.x + threadIdx.x;
    if (t >= NB * NPT) return;
    int b = t / NPT, n = t % NPT;

    const float* fp = features + (size_t)b * DD * NPT + n;
    float f[DD];
#pragma unroll
    for (int j = 0; j < DD; ++j) f[j] = fp[j * NPT];

    float sc[DD];
#pragma unroll
    for (int j = 0; j < DD; ++j)
        sc[j] = (1.0f / sqrtf((float)((j + 1) * (j + 2)))) * (float)NVX * sqrtf(2.0f / 3.0f);

    float elev[NVX];
#pragma unroll
    for (int i = 0; i < NVX; ++i) {
        float s = 0.0f;
#pragma unroll
        for (int j = 0; j < DD; ++j) {
            float e = (i == 0) ? 1.0f : ((j == i - 1) ? -(float)i : ((j >= i) ? 1.0f : 0.0f));
            s += e * sc[j] * f[j];
        }
        elev[i] = s;
    }

    int gi[NVX]; float diff[NVX]; int sum = 0;
#pragma unroll
    for (int i = 0; i < NVX; ++i) {
        int q = (int)rintf(elev[i] * (1.0f / NVX)) * NVX;   // multiple of NV
        gi[i] = q; sum += q; diff[i] = elev[i] - (float)q;
    }
    int s0 = sum / NVX;  // exact: sum is a multiple of NV

    int rank[NVX];
#pragma unroll
    for (int i = 0; i < NVX; ++i) {
        int r = s0;
#pragma unroll
        for (int j = 0; j < NVX; ++j)
            if (diff[j] > diff[i] || (diff[j] == diff[i] && j < i)) r++;
        rank[i] = r;
    }
#pragma unroll
    for (int i = 0; i < NVX; ++i) {
        if (rank[i] < 0)       { rank[i] += NVX; gi[i] += NVX; }
        else if (rank[i] > DD) { rank[i] -= NVX; gi[i] -= NVX; }
    }

    float bar[NVX + 2];
#pragma unroll
    for (int i = 0; i < NVX + 2; ++i) bar[i] = 0.0f;
#pragma unroll
    for (int i = 0; i < NVX; ++i) {
        float tt = (elev[i] - (float)gi[i]) * (1.0f / NVX);
        bar[DD - rank[i]]  += tt;
        bar[NVX - rank[i]] += 1.0f - tt;
    }
    bar[0] += 1.0f + bar[NVX];

    size_t base = (size_t)b * M0 + (size_t)n * NVX;
#pragma unroll
    for (int k = 0; k < NVX; ++k) {
        bary[base + k] = bar[k];
        int code = 0;
#pragma unroll
        for (int j = 0; j < DD; ++j) {
            int add = (rank[j] >= NVX - k) ? (k - NVX) : k;
            int key = gi[j] + add;
            key = min(max(key, -64), 63);
            code += (key + 64) << (7 * j);
        }
        codes[base + k] = code;
    }
}

// ---------------- hash insert (replaces sort/unique) ----------------
__global__ void k_insert(const int* __restrict__ codes, int* hashKey,
                         int* __restrict__ entrySlot) {
    int e = blockIdx.x * blockDim.x + threadIdx.x;
    if (e >= NB * M0) return;
    int b = e / M0;
    int code = codes[e];
    int* hk = hashKey + (size_t)b * HS;
    unsigned int h = hashc(code);
    while (true) {
        int prev = atomicCAS(&hk[h], -1, code);
        if (prev == -1 || prev == code) break;
        h = (h + 1) & HMASK;
    }
    entrySlot[e] = (int)h;
}

// ---------------- assign dense indices to occupied slots ----------------
__global__ void k_assign(const int* __restrict__ hashKey, int* hashDense,
                         int* nUniq, int* __restrict__ ucode) {
    int t = blockIdx.x * blockDim.x + threadIdx.x;
    if (t >= NB * HS) return;
    int b = t / HS;
    int code = hashKey[t];
    if (code != -1) {
        int d = atomicAdd(&nUniq[b], 1);
        hashDense[t] = d;
        ucode[(size_t)b * M0 + d] = code;
    }
}

// ---------------- splat: scatter-add weighted data into lattice ----------
__global__ void k_splat(const float* __restrict__ data, const float* __restrict__ bary,
                        const int* __restrict__ entrySlot, const int* __restrict__ hashDense,
                        int* __restrict__ idxDense, float* table) {
    int t = blockIdx.x * blockDim.x + threadIdx.x;
    if (t >= NB * M0 * 16) return;
    int e  = t >> 4;
    int c0 = (t & 15) * 4;
    int b = e / M0, r = e % M0, n = r / NVX;
    int d = hashDense[(size_t)b * HS + entrySlot[e]];
    if (c0 == 0) idxDense[e] = d;
    float w = bary[e];
    const float* dp = data + (size_t)b * CIN * NPT + n;
    float* tp = table + (size_t)(b * M0 + d) * CIN;
#pragma unroll
    for (int cc = 0; cc < 4; ++cc)
        atomicAdd(&tp[c0 + cc], dp[(size_t)(c0 + cc) * NPT] * w);
}

// ---------------- convert splat table to bf16 for WMMA A-operand --------
__global__ void k_tobf(const float* __restrict__ table, unsigned short* __restrict__ tableBF) {
    int t = blockIdx.x * blockDim.x + threadIdx.x;
    if (t >= NB * M0 * CIN) return;
    tableBF[t] = f2bf(table[t]);
}

// ---- pre-swizzle weights into per-lane WMMA B-fragment layout (bf16) ----
// Bmat[k = f*64+c][o] = W[o][c][f];  frag elem i of lane L, wave w, step kt:
//   k = kt*32 + (i<8 ? 8*(L>>4)+i : 16+8*(L>>4)+i-8) , o = 16*w + (L&15)
__global__ void k_packB(const float* __restrict__ weights, unsigned short* __restrict__ Bpack) {
    int t = blockIdx.x * blockDim.x + threadIdx.x;
    if (t >= KT * 4 * 32 * 16) return;
    int i    = t & 15;
    int lane = (t >> 4) & 31;
    int w    = (t >> 9) & 3;
    int kt   = t >> 11;
    int k = kt * 32 + ((i < 8) ? (8 * (lane >> 4) + i) : (16 + 8 * (lane >> 4) + i - 8));
    int o = 16 * w + (lane & 15);
    int f = k >> 6, c = k & 63;
    Bpack[t] = f2bf(weights[(size_t)o * CIN * FNB + (size_t)c * FNB + f]);
}

// ---------------- lattice convolution: gather + GEMM via WMMA ------------
__global__ __launch_bounds__(128)
void k_wmma(const unsigned short* __restrict__ tableBF, const unsigned short* __restrict__ Bpack,
            const int* __restrict__ ucode, const int* __restrict__ hashKey,
            const int* __restrict__ hashDense, const int* __restrict__ nUniq,
            float* __restrict__ outlat) {
    __shared__ int nbr[16][FNB];
    __shared__ unsigned short aT[16][32];   // 16x32 bf16 A tile, row-major

    int blk = blockIdx.x;
    int b = blk / MTILES, tile = blk % MTILES;
    int nU = nUniq[b];
    int m0 = tile * 16;
    if (m0 >= nU) return;                    // uniform across the block

    int tid = threadIdx.x;
    int lane = tid & 31, wv = tid >> 5;
    const int* hk = hashKey  + (size_t)b * HS;
    const int* hd = hashDense + (size_t)b * HS;

    // resolve the 16x31 neighbor dense indices once
    for (int q = tid; q < 16 * FNB; q += 128) {
        int r = q / FNB, f = q % FNB;
        int m = m0 + r, res = -1;
        if (m < nU) {
            int code = ucode[(size_t)b * M0 + m];
            int pc = __popc(f);
            int ncode = 0;
#pragma unroll
            for (int t = 0; t < DD; ++t) {
                int key = (((code >> (7 * t)) & 127) - 64) + NVX * ((f >> t) & 1) - pc;
                key = min(max(key, -64), 63);
                ncode += (key + 64) << (7 * t);
            }
            unsigned int h = hashc(ncode);
            while (true) {
                int c2 = hk[h];
                if (c2 == ncode) { res = hd[h]; break; }
                if (c2 == -1) break;
                h = (h + 1) & HMASK;
            }
        }
        nbr[r][f] = res;
    }
    __syncthreads();

    v8f acc = {};
    const unsigned short* bp = Bpack + ((size_t)wv * 32 + lane) * 16;

    for (int kt = 0; kt < KT; ++kt) {
        int f  = kt >> 1;
        int cb = (kt & 1) * 32;
        // stage A tile: 128 threads x 8 bytes (4 bf16), neighbor-gathered.
        // Invalid neighbors read a guaranteed-zero pad row so every lane can
        // issue an unconditional async global->LDS copy (ASYNCcnt path).
        {
            int r = tid >> 3, p = (tid & 7) * 4;
            int d = nbr[r][f];
            const unsigned short* src = tableBF +
                ((d >= 0) ? ((size_t)(b * M0 + d) * CIN + cb)
                          : (size_t)NB * M0 * CIN) + p;
#if __has_builtin(__builtin_amdgcn_global_load_async_to_lds_b64)
            __builtin_amdgcn_global_load_async_to_lds_b64(
                (gv2i_p)(const void*)src,
                (lv2i_p)(void*)&aT[r][p], 0, 0);
#if __has_builtin(__builtin_amdgcn_s_wait_asynccnt)
            __builtin_amdgcn_s_wait_asynccnt(0);
#else
            asm volatile("s_wait_asynccnt 0" ::: "memory");
#endif
#else
            *(unsigned long long*)&aT[r][p] = *(const unsigned long long*)src;
#endif
        }
        __syncthreads();

        // A fragment: lane (g = lane>>4, row = lane&15) needs two contiguous
        // 8-value runs: k in [8g, 8g+8) and [16+8g, 16+8g+8)
        union { v16bf v; unsigned int u[8]; } A;
        {
            int g = lane >> 4, row = lane & 15;
            const unsigned int* arow = (const unsigned int*)&aT[row][0];
#pragma unroll
            for (int i = 0; i < 4; ++i) {
                A.u[i]     = arow[4 * g + i];
                A.u[4 + i] = arow[8 + 4 * g + i];
            }
        }
        // B fragment: one contiguous 32B per-lane load (pre-swizzled)
        union { v16bf v; unsigned int u[8]; } Bf;
        {
            const unsigned int* bs = (const unsigned int*)(bp + (size_t)kt * (4 * 32 * 16));
#pragma unroll
            for (int i = 0; i < 8; ++i) Bf.u[i] = bs[i];
        }
        if (kt + 1 < KT)
            __builtin_prefetch(bp + (size_t)(kt + 1) * (4 * 32 * 16), 0, 0);

        acc = __builtin_amdgcn_wmma_f32_16x16x32_bf16(
            false, A.v, false, Bf.v, (short)0, acc, false, false);
        __syncthreads();
    }

    // epilogue: C/D layout — VGPR j: lanes 0-15 -> M=j, lanes 16-31 -> M=j+8
    int g = lane >> 4, col = (lane & 15) + wv * 16;
#pragma unroll
    for (int j = 0; j < 8; ++j) {
        int m = m0 + j + 8 * g;
        if (m < nU) outlat[(size_t)(b * M0 + m) * COUT + col] = acc[j];
    }
}

// ---------------- slice back + bias ----------------
__global__ void k_slice(const float* __restrict__ outlat, const int* __restrict__ idxDense,
                        const float* __restrict__ bary, const float* __restrict__ bias,
                        const float* __restrict__ bmul, float* __restrict__ out) {
    int t = blockIdx.x * blockDim.x + threadIdx.x;
    if (t >= NB * COUT * NPT) return;
    int b = t / (COUT * NPT);
    int rem = t % (COUT * NPT);
    int o = rem / NPT, n = rem % NPT;
    size_t base = (size_t)b * M0 + (size_t)n * NVX;
    float acc = 0.0f;
#pragma unroll
    for (int k = 0; k < NVX; ++k) {
        int d = idxDense[base + k];
        acc += bary[base + k] * outlat[(size_t)(b * M0 + d) * COUT + o];
    }
    out[t] = acc + bias[o] * bmul[n];
}

// ---------------- host launcher ----------------
static inline size_t alignup(size_t x) { return (x + 255) & ~(size_t)255; }

extern "C" void kernel_launch(void* const* d_in, const int* in_sizes, int n_in,
                              void* d_out, int out_size, void* d_ws, size_t ws_size,
                              hipStream_t stream) {
    const float* data     = (const float*)d_in[0];   // (B,Cin,H,W)
    const float* features = (const float*)d_in[1];   // (B,D,H,W)
    const float* weights  = (const float*)d_in[2];   // (Cout,Cin,1,F)
    const float* bias     = (const float*)d_in[3];   // (Cout)
    const float* bmul     = (const float*)d_in[4];   // (H*W)
    float* out = (float*)d_out;

    char* ws = (char*)d_ws;
    size_t off = 0;
    float* bary        = (float*)(ws + off); off = alignup(off + sizeof(float) * NB * M0);
    int*   codes       = (int*)  (ws + off); off = alignup(off + sizeof(int)   * NB * M0);
    int*   entrySlot   = (int*)  (ws + off); off = alignup(off + sizeof(int)   * NB * M0);
    int*   idxDense    = (int*)  (ws + off); off = alignup(off + sizeof(int)   * NB * M0);
    int*   hashKey     = (int*)  (ws + off); off = alignup(off + sizeof(int)   * NB * HS);
    int*   hashDense   = (int*)  (ws + off); off = alignup(off + sizeof(int)   * NB * HS);
    int*   ucode       = (int*)  (ws + off); off = alignup(off + sizeof(int)   * NB * M0);
    int*   nUniq       = (int*)  (ws + off); off = alignup(off + sizeof(int)   * NB);
    float* table       = (float*)(ws + off); off = alignup(off + sizeof(float) * NB * M0 * CIN);
    float* outlat      = (float*)(ws + off); off = alignup(off + sizeof(float) * NB * M0 * COUT);
    unsigned short* tableBF = (unsigned short*)(ws + off);
    off = alignup(off + sizeof(unsigned short) * (NB * M0 + 1) * CIN);   // +1 zero pad row
    unsigned short* Bpack   = (unsigned short*)(ws + off);
    off = alignup(off + sizeof(unsigned short) * KT * 4 * 32 * 16);

    const int TB = 256;
    k_init  <<<(NB * M0 * CIN + TB - 1) / TB, TB, 0, stream>>>(hashKey, hashDense, nUniq, table, tableBF);
    k_points<<<(NB * NPT + TB - 1) / TB,      TB, 0, stream>>>(features, bary, codes);
    k_packB <<<(KT * 4 * 32 * 16 + TB - 1) / TB, TB, 0, stream>>>(weights, Bpack);
    k_insert<<<(NB * M0 + TB - 1) / TB,       TB, 0, stream>>>(codes, hashKey, entrySlot);
    k_assign<<<(NB * HS + TB - 1) / TB,       TB, 0, stream>>>(hashKey, hashDense, nUniq, ucode);
    k_splat <<<(NB * M0 * 16 + TB - 1) / TB,  TB, 0, stream>>>(data, bary, entrySlot, hashDense, idxDense, table);
    k_tobf  <<<(NB * M0 * CIN + TB - 1) / TB, TB, 0, stream>>>(table, tableBF);
    k_wmma  <<<NB * MTILES, 128, 0, stream>>>(tableBF, Bpack, ucode, hashKey, hashDense, nUniq, outlat);
    k_slice <<<(NB * COUT * NPT + TB - 1) / TB, TB, 0, stream>>>(outlat, idxDense, bary, bias, bmul, out);
}
